// MultiHeadAttention_33998961115700
// MI455X (gfx1250) — matrix operand order repaired and checked by reading, not compiled, not run on hardware
//
#include <hip/hip_runtime.h>

// ---------------------------------------------------------------------------
// Types for CDNA5 WMMA (wave32, v_wmma_f32_16x16x32_bf16) + TDM descriptors
// ---------------------------------------------------------------------------
typedef __bf16 bf16_t;
typedef __attribute__((ext_vector_type(16))) __bf16 v16bf;
typedef __attribute__((ext_vector_type(8)))  __bf16 v8bf;
typedef __attribute__((ext_vector_type(8)))  float  v8f;
typedef __attribute__((ext_vector_type(4)))  unsigned int v4u;
typedef __attribute__((ext_vector_type(8)))  int v8i;
typedef __attribute__((ext_vector_type(4)))  int v4i;

static constexpr int Bd = 4, Td = 2048, Cd = 1024, Hd = 16, HDd = 64;

#if __has_builtin(__builtin_amdgcn_tensor_load_to_lds) && __has_builtin(__builtin_amdgcn_s_wait_tensorcnt)
#define USE_TDM 1
#else
#define USE_TDM 0
#endif

__device__ __forceinline__ v16bf cat16(v8bf lo, v8bf hi) {
  v16bf r;
#pragma unroll
  for (int i = 0; i < 8; ++i) { r[i] = lo[i]; r[i + 8] = hi[i]; }
  return r;
}

__device__ __forceinline__ v8f wmma_bf16(v16bf a, v16bf b, v8f c) {
  return __builtin_amdgcn_wmma_f32_16x16x32_bf16(false, a, false, b,
                                                 (short)0, c, false, false);
}

__device__ __forceinline__ void store_out(float* p, float v)  { *p = v; }
__device__ __forceinline__ void store_out(bf16_t* p, float v) { *p = (bf16_t)v; }

#if USE_TDM
// ---------------------------------------------------------------------------
// TDM: 2D tile load Global->LDS (bf16 elements), hardware LDS row padding.
// D# per CDNA5 ISA. tensor dims == tile dims
// (all shapes here divide exactly, so no OOB handling needed).
//   pad_interval: pad after 2^(v+1) DWORDs;  pad_amount: (v+1) DWORDs.
// This toolchain exposes the 6-arg builtin:
//   (uint32x4 g0, int32x8 g1, int32x4 g2, int32x4 g3, int32x8, i32 cpol)
// ---------------------------------------------------------------------------
__device__ __forceinline__ void tdm_load_2d_bf16(
    unsigned int lds_addr, const bf16_t* gaddr,
    unsigned int tile_d0, unsigned int tile_d1, unsigned int stride_d0,
    unsigned int pad_interval, unsigned int pad_amount) {
  unsigned long long ga = (unsigned long long)(uintptr_t)gaddr;
  v4u g0;
  g0[0] = 1u;                                   // count=1, user mode
  g0[1] = lds_addr;                             // LDS byte address
  g0[2] = (unsigned int)ga;                     // global_addr[31:0]
  g0[3] = (unsigned int)((ga >> 32) & 0x1FFFFFFu) | (2u << 30);  // [56:32], type=2
  v8i g1;
  g1[0] = (int)((1u << 16) |                    // data_size = 2 bytes
                (1u << 20) |                    // pad_enable
                (pad_interval << 22) | (pad_amount << 25));
  g1[1] = (int)((tile_d0 & 0xFFFFu) << 16);                         // tensor_dim0 lo
  g1[2] = (int)(((tile_d0 >> 16) & 0xFFFFu) | ((tile_d1 & 0xFFFFu) << 16));
  g1[3] = (int)(((tile_d1 >> 16) & 0xFFFFu) | ((tile_d0 & 0xFFFFu) << 16)); // tile_dim0
  g1[4] = (int)(tile_d1 & 0xFFFFu);             // tile_dim1, tile_dim2=0
  g1[5] = (int)stride_d0;                       // tensor_dim0_stride[31:0] (elems)
  g1[6] = 0;
  g1[7] = 0;
  v4i z4 = {};
  v8i z8 = {};
  __builtin_amdgcn_tensor_load_to_lds(g0, g1, z4, z4, z8, 0);
}
#endif

// ---------------------------------------------------------------------------
// Kernel 0a: f32 -> bf16 conversion (grid-stride)
// ---------------------------------------------------------------------------
__global__ void f32_to_bf16(const float* __restrict__ in,
                            bf16_t* __restrict__ out, long n) {
  long i = (long)blockIdx.x * blockDim.x + threadIdx.x;
  long stride = (long)gridDim.x * blockDim.x;
  for (; i < n; i += stride) out[i] = (bf16_t)in[i];
}

// ---------------------------------------------------------------------------
// Kernel 0b: f32[K][N] -> bf16[N][K] tiled transpose (weights pre-transpose)
// ---------------------------------------------------------------------------
__global__ __launch_bounds__(256) void transpose_to_bf16(
    const float* __restrict__ in, bf16_t* __restrict__ out, int K, int N) {
  __shared__ float tile[32][33];
  const int n0 = blockIdx.x * 32, k0 = blockIdx.y * 32;
  const int tx = threadIdx.x & 31, ty = threadIdx.x >> 5;  // 32 x 8
#pragma unroll
  for (int j = 0; j < 4; ++j)
    tile[ty + j * 8][tx] = in[(size_t)(k0 + ty + j * 8) * N + n0 + tx];
  __syncthreads();
#pragma unroll
  for (int j = 0; j < 4; ++j)
    out[(size_t)(n0 + ty + j * 8) * K + k0 + tx] = (bf16_t)tile[tx][ty + j * 8];
}

// ---------------------------------------------------------------------------
// Kernel 1/3: bf16 GEMM, C[M,N] = A[M,K] * Bt[N,K]^T, f32 accumulate.
// Block tile 128x128, K-step 32, 8 waves, wave tile 32x64.
// Both operands staged as 128x32 contiguous-row tiles -> TDM double-buffered.
// LDS row stride 40 bf16 (80 B) provided by TDM pad (16 DW interval, 4 DW pad).
// ---------------------------------------------------------------------------
template <typename OutT>
__global__ __launch_bounds__(256) void gemm_bf16_wmma(
    const bf16_t* __restrict__ A, const bf16_t* __restrict__ Bt,
    OutT* __restrict__ Cout, int M, int N, int K) {
  __shared__ alignas(16) bf16_t As[2][128 * 40];
  __shared__ alignas(16) bf16_t Bs[2][128 * 40];

  const int tid  = threadIdx.x;
  const int wave = tid >> 5;
  const int lane = tid & 31;
  const int half = lane >> 4;
  const int lr   = lane & 15;
  const int wr   = wave & 3;   // 4 wave-rows * 32
  const int wc   = wave >> 2;  // 2 wave-cols * 64
  const int m0   = blockIdx.y * 128;
  const int n0   = blockIdx.x * 128;

  v8f acc[2][4] = {};
  const int niter = K >> 5;

#if USE_TDM
  const unsigned int asb[2] = {(unsigned int)(uintptr_t)&As[0][0],
                               (unsigned int)(uintptr_t)&As[1][0]};
  const unsigned int bsb[2] = {(unsigned int)(uintptr_t)&Bs[0][0],
                               (unsigned int)(uintptr_t)&Bs[1][0]};
  if (wave == 0) {  // prologue: fill buffer 0
    tdm_load_2d_bf16(asb[0], A  + (size_t)m0 * K, 32, 128, (unsigned)K, 3, 3);
    tdm_load_2d_bf16(bsb[0], Bt + (size_t)n0 * K, 32, 128, (unsigned)K, 3, 3);
  }
#endif

  for (int i = 0; i < niter; ++i) {
    const int buf = i & 1;
#if USE_TDM
    if (wave == 0) {
      if (i + 1 < niter) {  // async-prefetch next tile pair into other buffer
        const int k0n = (i + 1) << 5;
        tdm_load_2d_bf16(asb[buf ^ 1], A  + (size_t)m0 * K + k0n, 32, 128, (unsigned)K, 3, 3);
        tdm_load_2d_bf16(bsb[buf ^ 1], Bt + (size_t)n0 * K + k0n, 32, 128, (unsigned)K, 3, 3);
        __builtin_amdgcn_s_wait_tensorcnt(2);  // current pair done
      } else {
        __builtin_amdgcn_s_wait_tensorcnt(0);
      }
    }
    __syncthreads();  // tile visible to all waves
#else
    const int k0 = i << 5;
    __syncthreads();
#pragma unroll
    for (int it = 0; it < 2; ++it) {
      int s = it * 256 + tid;            // 0..511
      int row = s >> 2, seg = s & 3;     // 4 segments of 8 bf16
      v8bf ta = *(const v8bf*)(A + (size_t)(m0 + row) * K + k0 + seg * 8);
      *(v8bf*)&As[buf][row * 40 + seg * 8] = ta;
      v8bf tb = *(const v8bf*)(Bt + (size_t)(n0 + row) * K + k0 + seg * 8);
      *(v8bf*)&Bs[buf][row * 40 + seg * 8] = tb;
    }
    __syncthreads();
#endif

    // ---- fragments ----
    // A frag (16x32 bf16): half 0 -> K 0..15, half 1 -> K 16..31 (contiguous)
    // B frag (32x16 bf16): half 0 -> K 0..15, half 1 -> K 16..31 (contiguous)
    v16bf af[2], bfr[4];
#pragma unroll
    for (int mt = 0; mt < 2; ++mt) {
      int row = wr * 32 + mt * 16 + lr;
      af[mt] = cat16(*(const v8bf*)&As[buf][row * 40 + half * 16],
                     *(const v8bf*)&As[buf][row * 40 + half * 16 + 8]);
    }
#pragma unroll
    for (int nt = 0; nt < 4; ++nt) {
      int col = wc * 64 + nt * 16 + lr;
      bfr[nt] = cat16(*(const v8bf*)&Bs[buf][col * 40 + half * 16],
                      *(const v8bf*)&Bs[buf][col * 40 + half * 16 + 8]);
    }
#pragma unroll
    for (int mt = 0; mt < 2; ++mt)
#pragma unroll
      for (int nt = 0; nt < 4; ++nt)
        acc[mt][nt] = wmma_bf16(af[mt], bfr[nt], acc[mt][nt]);

    __syncthreads();  // free this buffer for the TDM issued next iteration
  }

  // ---- store: C/D layout: VGPR r -> row r + 8*half, col = lane%16 ----
#pragma unroll
  for (int mt = 0; mt < 2; ++mt)
#pragma unroll
    for (int nt = 0; nt < 4; ++nt)
#pragma unroll
      for (int r = 0; r < 8; ++r) {
        int row = m0 + wr * 32 + mt * 16 + r + 8 * half;
        int col = n0 + wc * 64 + nt * 16 + lr;
        store_out(Cout + (size_t)row * N + col, acc[mt][nt][r]);
      }
}

// ---------------------------------------------------------------------------
// Kernel 2: flash attention (causal), bf16 WMMA + f32 online softmax.
// One block = one (b,h) x 128 q-rows. 8 waves; wave owns 16 q rows.
// K-block staged by TDM (tile 64x64, stride 3C, LDS padded to 72 bf16/row);
// V staged transposed on the VALU path, overlapping the TDM transfer.
// ---------------------------------------------------------------------------
__global__ __launch_bounds__(256) void flash_attn_wmma(
    const bf16_t* __restrict__ qkv, bf16_t* __restrict__ y) {
  __shared__ alignas(16) bf16_t Ks[64 * 72];       // [key][hd], stride 72
  __shared__ alignas(16) bf16_t Vt[64 * 72];       // [hd][key], stride 72
  __shared__ alignas(16) bf16_t Ps[8 * 16 * 72];   // per-wave P tile [row][key]

  const int tid  = threadIdx.x;
  const int wave = tid >> 5;
  const int lane = tid & 31;
  const int half = lane >> 4;
  const int lr   = lane & 15;

  const int nqb = Td / 128;                 // 16
  const int qb  = blockIdx.x % nqb;
  const int bh  = blockIdx.x / nqb;
  const int b   = bh / Hd;
  const int h   = bh % Hd;

  const size_t row_stride = (size_t)3 * Cd;
  const bf16_t* qp = qkv + (size_t)b * Td * 3 * Cd + h * HDd;
  const bf16_t* kp = qp + Cd;
  const bf16_t* vp = qp + 2 * Cd;

  // ---- Q fragments for this wave's 16 rows (2 K-windows over HD=64) ----
  const int qrow = qb * 128 + wave * 16 + lr;
  v16bf qf[2];
#pragma unroll
  for (int s = 0; s < 2; ++s) {
    const bf16_t* qr = qp + (size_t)qrow * row_stride + s * 32;
    qf[s] = cat16(*(const v8bf*)(qr + half * 16),
                  *(const v8bf*)(qr + half * 16 + 8));
  }

  v8f o[4] = {};
  float mrow[8], lsum[8];
#pragma unroll
  for (int r = 0; r < 8; ++r) { mrow[r] = -1e30f; lsum[r] = 0.0f; }

  bf16_t* ps = &Ps[wave * 16 * 72];
  const int qg0 = qb * 128 + wave * 16 + 8 * half;  // + r = global q row
#if USE_TDM
  const unsigned int ks_addr = (unsigned int)(uintptr_t)&Ks[0];
#endif

  const int kb_end = 2 * qb + 2;  // causal: keys up to qb*128+127
  for (int kb = 0; kb < kb_end; ++kb) {
    __syncthreads();  // previous block's Ks/Vt reads complete
#if USE_TDM
    if (wave == 0)    // DMA the K block while all waves stage V below
      tdm_load_2d_bf16(ks_addr, kp + (size_t)(kb * 64) * row_stride,
                       64, 64, (unsigned)row_stride, 4, 3);
#else
#pragma unroll
    for (int it = 0; it < 2; ++it) {
      int s = it * 256 + tid;
      int key = s >> 3, seg = s & 7;
      v8bf kt = *(const v8bf*)(kp + (size_t)(kb * 64 + key) * row_stride + seg * 8);
      *(v8bf*)&Ks[key * 72 + seg * 8] = kt;
    }
#endif
    // ---- stage V transposed: Vt[hd][key] ----
#pragma unroll
    for (int it = 0; it < 2; ++it) {
      int s = it * 256 + tid;           // 0..511
      int key = s >> 3, seg = s & 7;    // 8 segments of 8 bf16
      v8bf vt = *(const v8bf*)(vp + (size_t)(kb * 64 + key) * row_stride + seg * 8);
#pragma unroll
      for (int i = 0; i < 8; ++i) Vt[(seg * 8 + i) * 72 + key] = vt[i];
    }
#if USE_TDM
    if (wave == 0) __builtin_amdgcn_s_wait_tensorcnt(0);
#endif
    __syncthreads();

    // ---- S = q @ k^T (wave: 16 rows x 64 keys -> 4 accums) ----
    v8f sa[4];
#pragma unroll
    for (int nt = 0; nt < 4; ++nt) {
      int col = nt * 16 + lr;  // key within block
      v16bf kf0 = cat16(*(const v8bf*)&Ks[col * 72 + half * 16],
                        *(const v8bf*)&Ks[col * 72 + half * 16 + 8]);
      v16bf kf1 = cat16(*(const v8bf*)&Ks[col * 72 + 32 + half * 16],
                        *(const v8bf*)&Ks[col * 72 + 32 + half * 16 + 8]);
      v8f z = {};
      z = wmma_bf16(qf[0], kf0, z);
      z = wmma_bf16(qf[1], kf1, z);
      sa[nt] = z;
    }

    // ---- causal mask + online softmax (rows live per lane-half) ----
#pragma unroll
    for (int r = 0; r < 8; ++r) {
      float mx = -1e30f;
#pragma unroll
      for (int nt = 0; nt < 4; ++nt) {
        int kg = kb * 64 + nt * 16 + lr;
        float v = sa[nt][r] * 0.125f;             // 1/sqrt(64)
        v = (kg > qg0 + r) ? -1e30f : v;
        sa[nt][r] = v;
        mx = fmaxf(mx, v);
      }
#pragma unroll
      for (int off = 1; off < 16; off <<= 1)
        mx = fmaxf(mx, __shfl_xor(mx, off, 32));
      float mnew  = fmaxf(mrow[r], mx);
      float alpha = __expf(mrow[r] - mnew);
      float rs = 0.0f;
#pragma unroll
      for (int nt = 0; nt < 4; ++nt) {
        float p = __expf(sa[nt][r] - mnew);
        sa[nt][r] = p;
        rs += p;
      }
#pragma unroll
      for (int off = 1; off < 16; off <<= 1)
        rs += __shfl_xor(rs, off, 32);
      lsum[r] = lsum[r] * alpha + rs;
      mrow[r] = mnew;
#pragma unroll
      for (int nt = 0; nt < 4; ++nt) o[nt][r] *= alpha;
    }

    // ---- spill P (C-layout) to wave-private LDS, reload as A-fragments ----
#pragma unroll
    for (int nt = 0; nt < 4; ++nt)
#pragma unroll
      for (int r = 0; r < 8; ++r)
        ps[(r + 8 * half) * 72 + nt * 16 + lr] = (bf16_t)sa[nt][r];

    v16bf pf[2];
#pragma unroll
    for (int s = 0; s < 2; ++s)
      pf[s] = cat16(*(const v8bf*)&ps[lr * 72 + s * 32 + half * 16],
                    *(const v8bf*)&ps[lr * 72 + s * 32 + half * 16 + 8]);

    // ---- O += P @ V  (B-frag from transposed V) ----
#pragma unroll
    for (int nt = 0; nt < 4; ++nt) {
      int n = nt * 16 + lr;  // hd column
      v16bf vf0 = cat16(*(const v8bf*)&Vt[n * 72 + half * 16],
                        *(const v8bf*)&Vt[n * 72 + half * 16 + 8]);
      v16bf vf1 = cat16(*(const v8bf*)&Vt[n * 72 + 32 + half * 16],
                        *(const v8bf*)&Vt[n * 72 + 32 + half * 16 + 8]);
      o[nt] = wmma_bf16(pf[0], vf0, o[nt]);
      o[nt] = wmma_bf16(pf[1], vf1, o[nt]);
    }
  }

  // ---- epilogue: normalize and write y[b, t, h*64 + hd] (bf16) ----
#pragma unroll
  for (int r = 0; r < 8; ++r) {
    float inv = 1.0f / lsum[r];
    int qg = qg0 + r;
    size_t base = ((size_t)b * Td + qg) * Cd + h * HDd;
#pragma unroll
    for (int nt = 0; nt < 4; ++nt)
      y[base + nt * 16 + lr] = (bf16_t)(o[nt][r] * inv);
  }
}

// ---------------------------------------------------------------------------
// Launch: convert/transpose -> qkv GEMM -> flash attention -> proj GEMM
// ---------------------------------------------------------------------------
extern "C" void kernel_launch(void* const* d_in, const int* in_sizes, int n_in,
                              void* d_out, int out_size, void* d_ws, size_t ws_size,
                              hipStream_t stream) {
  const float* x      = (const float*)d_in[0];
  const float* w_attn = (const float*)d_in[1];
  const float* w_proj = (const float*)d_in[2];
  float* out = (float*)d_out;

  const size_t nx   = (size_t)Bd * Td * Cd;       // 8,388,608
  const size_t nwa  = (size_t)Cd * 3 * Cd;        // 3,145,728
  const size_t nwp  = (size_t)Cd * Cd;            // 1,048,576
  const size_t nqkv = (size_t)Bd * Td * 3 * Cd;   // 25,165,824

  bf16_t* xb   = (bf16_t*)d_ws;
  bf16_t* wat  = xb + nx;      // w_attn transposed: [3C][C]
  bf16_t* wpt  = wat + nwa;    // w_proj transposed: [C][C]
  bf16_t* qkvb = wpt + nwp;
  bf16_t* yb   = qkvb + nqkv;  // total ~92.3 MB of workspace

  // 0) f32 -> bf16 (x) and f32 -> bf16 transposed (weights)
  f32_to_bf16<<<4096, 256, 0, stream>>>(x, xb, (long)nx);
  {
    dim3 g(3 * Cd / 32, Cd / 32);   // (96, 32)
    transpose_to_bf16<<<g, 256, 0, stream>>>(w_attn, wat, Cd, 3 * Cd);
  }
  {
    dim3 g(Cd / 32, Cd / 32);       // (32, 32)
    transpose_to_bf16<<<g, 256, 0, stream>>>(w_proj, wpt, Cd, Cd);
  }

  // 1) qkv = x @ w_attn : [8192 x 1024] * [1024 x 3072]
  {
    dim3 grid(3 * Cd / 128, (Bd * Td) / 128);  // (24, 64)
    gemm_bf16_wmma<bf16_t><<<grid, 256, 0, stream>>>(
        xb, wat, qkvb, Bd * Td, 3 * Cd, Cd);
  }

  // 2) flash attention -> yb (bf16, [B, T, C] with heads interleaved)
  flash_attn_wmma<<<(Bd * Hd) * (Td / 128), 256, 0, stream>>>(qkvb, yb);

  // 3) out = y @ w_proj : [8192 x 1024] * [1024 x 1024], f32 output
  {
    dim3 grid(Cd / 128, (Bd * Td) / 128);  // (8, 64)
    gemm_bf16_wmma<float><<<grid, 256, 0, stream>>>(
        yb, wpt, out, Bd * Td, Cd, Cd);
  }
}
